// AGGCN_69578470195674
// MI455X (gfx1250) — matrix operand order repaired and checked by reading, not compiled, not run on hardware
//
#include <hip/hip_runtime.h>
#include <hip/hip_bf16.h>
#include <cstdint>

#define NNODE 512
#define TSTEP 12
#define BBATCH 64
#define DEMB 16
#define HDIM 128
#define DIN_C 2

typedef __bf16 bf16_t;
typedef __attribute__((ext_vector_type(16))) __bf16 v16bf;
typedef __attribute__((ext_vector_type(8)))  __bf16 v8bf;
typedef __attribute__((ext_vector_type(8)))  float  v8f;

// ---------------- WMMA fragment helpers (wave32, bf16 16x16x32) ----------------
// A (16x32, row-major src): lane r=l%16 holds row r; elems 0..15 = contiguous
// K[h*16 .. h*16+16), h = l/16 (matches B fragment order).
static __device__ __forceinline__ v16bf load_fragA(const bf16_t* __restrict__ base, int stride) {
  const int lane = threadIdx.x & 31;
  const int r = lane & 15, h = lane >> 4;
  v8bf lo = *(const v8bf*)(base + (size_t)r * stride + h * 16);
  v8bf hi = *(const v8bf*)(base + (size_t)r * stride + h * 16 + 8);
  v16bf f;
#pragma unroll
  for (int e = 0; e < 8; ++e) { f[e] = lo[e]; f[e + 8] = hi[e]; }
  return f;
}
// B (32x16) loaded from B^T (row-major, row = output column): lane c=l%16 holds
// column c, contiguous K[h*16 .. h*16+16) (same order as A).
static __device__ __forceinline__ v16bf load_fragB(const bf16_t* __restrict__ base, int stride) {
  const int lane = threadIdx.x & 31;
  const int c = lane & 15, h = lane >> 4;
  return *(const v16bf*)(base + (size_t)c * stride + h * 16);
}
static __device__ __forceinline__ v8f wmma_bf16(v16bf a, v16bf b, v8f c) {
  return __builtin_amdgcn_wmma_f32_16x16x32_bf16(false, a, false, b, (short)0, c, false, false);
}
// load a 64x32 A-panel (4 frags) + 32x32 B-panel (2 frags) at K-offset kc
static __device__ __forceinline__ void load_tileAB(const bf16_t* __restrict__ A,
                                                   const bf16_t* __restrict__ BT,
                                                   int Kdim, int m0, int j0, int kc,
                                                   v16bf a[4], v16bf b[2]) {
  b[0] = load_fragB(BT + (size_t)j0 * Kdim + kc, Kdim);
  b[1] = load_fragB(BT + (size_t)(j0 + 16) * Kdim + kc, Kdim);
#pragma unroll
  for (int r = 0; r < 4; ++r)
    a[r] = load_fragA(A + (size_t)(m0 + r * 16) * Kdim + kc, Kdim);
}

// ---------------- small prep kernels ----------------
__global__ void k_neln(const float* __restrict__ nemb, const float* __restrict__ temb,
                       const float* __restrict__ gg, const float* __restrict__ gbeta,
                       const float* __restrict__ ug, const float* __restrict__ ubeta,
                       float* __restrict__ neG, float* __restrict__ neU) {
  int idx = blockIdx.x * blockDim.x + threadIdx.x;
  if (idx >= TSTEP * NNODE) return;
  int n = idx % NNODE, t = idx / NNODE;
  float raw[DEMB]; float m = 0.f;
#pragma unroll
  for (int d = 0; d < DEMB; ++d) { raw[d] = nemb[n*DEMB+d] + temb[t*DEMB+d]; m += raw[d]; }
  m *= (1.f / DEMB);
  float v = 0.f;
#pragma unroll
  for (int d = 0; d < DEMB; ++d) { float x = raw[d] - m; v += x * x; }
  v *= (1.f / DEMB);
  float inv = rsqrtf(v + 1e-12f);
#pragma unroll
  for (int d = 0; d < DEMB; ++d) {
    float x = (raw[d] - m) * inv;
    neG[(size_t)idx*DEMB + d] = x * gg[d] + gbeta[d];
    neU[(size_t)idx*DEMB + d] = x * ug[d] + ubeta[d];
  }
}

__global__ void k_bias(const float* __restrict__ ne, const float* __restrict__ Wb,
                       float* __restrict__ outb, int w) {
  size_t idx = (size_t)blockIdx.x * blockDim.x + threadIdx.x;
  if (idx >= (size_t)TSTEP * NNODE * w) return;
  int o = (int)(idx % w); size_t tn = idx / w;
  float s = 0.f;
#pragma unroll
  for (int d = 0; d < DEMB; ++d) s += ne[tn*DEMB + d] * Wb[(size_t)d*w + o];
  outb[idx] = s;
}

// Wt[o, (d*3+k)*cin_p + i] = W[d,k,i,o]  (bf16, zero pad i>=cin)
__global__ void k_wprep(const float* __restrict__ W, bf16_t* __restrict__ Wt,
                        int cin, int cin_p, int outw) {
  size_t tot = (size_t)outw * DEMB * 3 * cin_p;
  size_t idx = (size_t)blockIdx.x * blockDim.x + threadIdx.x;
  if (idx >= tot) return;
  int colsz = DEMB * 3 * cin_p;
  int col = (int)(idx % colsz); int o = (int)(idx / colsz);
  int i = col % cin_p; int dk = col / cin_p; int k = dk % 3; int d = dk / 3;
  float v = (i < cin) ? W[(((size_t)d*3 + k)*cin + i)*outw + o] : 0.f;
  Wt[idx] = (bf16_t)v;
}

// row-softmax of ne@ne^T for one (t,gate,n); writes S and S^T (bf16)
__global__ void k_softmax(const float* __restrict__ neG, const float* __restrict__ neU,
                          bf16_t* __restrict__ S, bf16_t* __restrict__ St) {
  int bid = blockIdx.x;
  int n = bid % NNODE; int tg = bid / NNODE; int g = tg & 1; int t = tg >> 1;
  const float* ne = (g ? neU : neG) + (size_t)t * NNODE * DEMB;
  __shared__ float row[NNODE];
  __shared__ float red[128];
  float p[DEMB];
#pragma unroll
  for (int d = 0; d < DEMB; ++d) p[d] = ne[(size_t)n*DEMB + d];
  int tid = threadIdx.x;
  float lmax = -1e30f;
  for (int m = tid; m < NNODE; m += 128) {
    float s = 0.f;
#pragma unroll
    for (int d = 0; d < DEMB; ++d) s += p[d] * ne[(size_t)m*DEMB + d];
    row[m] = s; lmax = fmaxf(lmax, s);
  }
  red[tid] = lmax; __syncthreads();
  for (int st = 64; st > 0; st >>= 1) { if (tid < st) red[tid] = fmaxf(red[tid], red[tid+st]); __syncthreads(); }
  float mx = red[0]; __syncthreads();
  float lsum = 0.f;
  for (int m = tid; m < NNODE; m += 128) { float e = __expf(row[m] - mx); row[m] = e; lsum += e; }
  red[tid] = lsum; __syncthreads();
  for (int st = 64; st > 0; st >>= 1) { if (tid < st) red[tid] += red[tid+st]; __syncthreads(); }
  float inv = 1.f / red[0];
  size_t base = (size_t)tg * NNODE * NNODE;
  for (int m = tid; m < NNODE; m += 128) {
    bf16_t v = (bf16_t)(row[m] * inv);
    S [base + (size_t)n*NNODE + m] = v;
    St[base + (size_t)m*NNODE + n] = v;
  }
}

// ---------------- WMMA GEMMs ----------------
// T2 = 2*S@S - I   (A=S, B^T=S^T, both 512x512 bf16), grid.z = t*2+gate
// 64x256 block tile, 8 waves, 4x2 subtiles/wave; K-loop unrolled 2x with
// explicit ping-pong buffers (no register rotation copies).
__global__ __launch_bounds__(256) void k_t2(const bf16_t* __restrict__ Sall,
                                            const bf16_t* __restrict__ Stall,
                                            bf16_t* __restrict__ T2all) {
  size_t zoff = (size_t)blockIdx.z * NNODE * NNODE;
  const bf16_t* A = Sall + zoff; const bf16_t* BT = Stall + zoff;
  bf16_t* C = T2all + zoff;
  int m0 = blockIdx.x * 64;
  int j0 = blockIdx.y * 256 + (threadIdx.x >> 5) * 32;
  v8f acc[4][2] = {};
  v16bf a0[4], b0[2], a1[4], b1[2];
  load_tileAB(A, BT, NNODE, m0, j0, 0, a0, b0);
  for (int kc = 0; kc < NNODE; kc += 64) {
    load_tileAB(A, BT, NNODE, m0, j0, kc + 32, a1, b1);
#pragma unroll
    for (int r = 0; r < 4; ++r) {
      acc[r][0] = wmma_bf16(a0[r], b0[0], acc[r][0]);
      acc[r][1] = wmma_bf16(a0[r], b0[1], acc[r][1]);
    }
    if (kc + 64 < NNODE) load_tileAB(A, BT, NNODE, m0, j0, kc + 64, a0, b0);
#pragma unroll
    for (int r = 0; r < 4; ++r) {
      acc[r][0] = wmma_bf16(a1[r], b1[0], acc[r][0]);
      acc[r][1] = wmma_bf16(a1[r], b1[1], acc[r][1]);
    }
  }
  int lane = threadIdx.x & 31;
  int rb = m0 + ((lane >> 4) << 3);
#pragma unroll
  for (int cc = 0; cc < 2; ++cc) {
    int col = j0 + cc * 16 + (lane & 15);
#pragma unroll
    for (int r = 0; r < 4; ++r)
#pragma unroll
      for (int jj = 0; jj < 8; ++jj) {
        int rowi = rb + r*16 + jj;
        float v = 2.f * acc[r][cc][jj] - (rowi == col ? 1.f : 0.f);
        C[(size_t)rowi * NNODE + col] = (bf16_t)v;
      }
  }
}

// Y_z = A_z @ X  (A row-major MxK, X^T row-major Ncols x K); z=0 -> A0 (S), z=1 -> A1 (T2)
__global__ __launch_bounds__(256) void k_gemm_ab(const bf16_t* __restrict__ A0,
                                                 const bf16_t* __restrict__ A1,
                                                 const bf16_t* __restrict__ BT,
                                                 bf16_t* __restrict__ Cout,
                                                 int Kdim, int Ncols) {
  const bf16_t* A = blockIdx.z ? A1 : A0;
  bf16_t* C = Cout + (size_t)blockIdx.z * (size_t)NNODE * Ncols;
  int m0 = blockIdx.x * 64;
  int j0 = blockIdx.y * 256 + (threadIdx.x >> 5) * 32;
  v8f acc[4][2] = {};
  v16bf a0[4], b0[2], a1[4], b1[2];
  load_tileAB(A, BT, Kdim, m0, j0, 0, a0, b0);
  for (int kc = 0; kc < Kdim; kc += 64) {   // Kdim = 512 (multiple of 64)
    load_tileAB(A, BT, Kdim, m0, j0, kc + 32, a1, b1);
#pragma unroll
    for (int r = 0; r < 4; ++r) {
      acc[r][0] = wmma_bf16(a0[r], b0[0], acc[r][0]);
      acc[r][1] = wmma_bf16(a0[r], b0[1], acc[r][1]);
    }
    if (kc + 64 < Kdim) load_tileAB(A, BT, Kdim, m0, j0, kc + 64, a0, b0);
#pragma unroll
    for (int r = 0; r < 4; ++r) {
      acc[r][0] = wmma_bf16(a1[r], b1[0], acc[r][0]);
      acc[r][1] = wmma_bf16(a1[r], b1[1], acc[r][1]);
    }
  }
  int lane = threadIdx.x & 31;
  int rb = m0 + ((lane >> 4) << 3);
#pragma unroll
  for (int cc = 0; cc < 2; ++cc) {
    int col = j0 + cc * 16 + (lane & 15);
#pragma unroll
    for (int r = 0; r < 4; ++r)
#pragma unroll
      for (int jj = 0; jj < 8; ++jj)
        C[(size_t)(rb + r*16 + jj) * Ncols + col] = (bf16_t)acc[r][cc][jj];
  }
}

// Build one 32K x 128o chunk of the node-effective weight in LDS:
// wbuf[o_l*32 + kk] = sum_d s[d] * Wt[o0+o_l, (d*3+k)*cin_p + i0 + kk]
static __device__ __forceinline__ void build_weff(const bf16_t* __restrict__ Wt, int o0,
                                                  int wstride, int cin_p, int k, int i0,
                                                  const float (&s)[DEMB], int tid,
                                                  bf16_t* wbuf) {
#pragma unroll
  for (int rep = 0; rep < 2; ++rep) {
    int tt = tid + (rep << 8);
    int o_l = tt >> 2, kg = tt & 3;
    const bf16_t* wp = Wt + (size_t)(o0 + o_l) * wstride + (size_t)k * cin_p + i0 + kg * 8;
    float accw[8] = {};
    for (int d = 0; d < DEMB; ++d) {
      v8bf w8 = *(const v8bf*)(wp + (size_t)d * 3 * cin_p);
      float sd = s[d];
#pragma unroll
      for (int e = 0; e < 8; ++e) accw[e] += sd * (float)w8[e];
    }
    v8bf wout;
#pragma unroll
    for (int e = 0; e < 8; ++e) wout[e] = (bf16_t)accw[e];
    *(v8bf*)(&wbuf[o_l * 32 + kg * 8]) = wout;
  }
}

// Node-conditioned contraction:
// Out[n,b,o] = bias[n,o] + sum_{k,i} Xg_k[n,b,i] * (sum_d ne[n,d] * W[d,k,i,o])
// One block per (n, o-group of 128). Effective weight chunks (8 KB bf16) are
// built into a double-buffered LDS staging area (one barrier per chunk), so the
// next chunk's global weight reads + VALU combine overlap the current chunk's
// ds_load B-fragments + WMMA. Keeps WMMA K at 3*cin_p (no d-axis inflation).
__global__ __launch_bounds__(256) void k_gemm2(const bf16_t* __restrict__ Xg0,
                                               const bf16_t* __restrict__ Y,
                                               const bf16_t* __restrict__ Wt,
                                               const float* __restrict__ ne_t,
                                               const float* __restrict__ bias_t,
                                               float* __restrict__ Out,
                                               int cin_p, int outw) {
  __shared__ __align__(32) bf16_t weff[2][128 * 32];   // double-buffered [o_local][kk]
  int n = blockIdx.x;
  int o0 = blockIdx.y * 128;
  int wv = threadIdx.x >> 5;
  int tid = threadIdx.x;
  const bf16_t* P0 = Xg0 + (size_t)n * BBATCH * cin_p;
  const bf16_t* P1 = Y   + (size_t)n * BBATCH * cin_p;
  const bf16_t* P2 = Y   + (size_t)(NNODE + n) * BBATCH * cin_p;
  float s[DEMB];
#pragma unroll
  for (int d = 0; d < DEMB; ++d) s[d] = ne_t[(size_t)n*DEMB + d];
  const int nkc = cin_p >> 5;
  const int nch = 3 * nkc;
  const int wstride = DEMB * 3 * cin_p;
  const int lane = tid & 31;
  const int c16 = lane & 15, hh = lane >> 4;
  v8f acc[4] = {};
  build_weff(Wt, o0, wstride, cin_p, 0, 0, s, tid, weff[0]);
  int kcur = 0, icur = 0;
  for (int c = 0; c < nch; ++c) {
    int i0 = icur << 5;
    const bf16_t* Pk = (kcur == 0) ? P0 : (kcur == 1) ? P1 : P2;
    // issue A-fragment loads early (overlap with the barrier + next build)
    v16bf a[4];
#pragma unroll
    for (int r = 0; r < 4; ++r) a[r] = load_fragA(Pk + (size_t)(r*16) * cin_p + i0, cin_p);
    __syncthreads();   // publishes chunk c; protects buffer consumed at c-1
    int icn = icur + 1, kn = kcur;
    if (icn == nkc) { icn = 0; kn = kcur + 1; }
    if (c + 1 < nch)
      build_weff(Wt, o0, wstride, cin_p, kn, icn << 5, s, tid, weff[(c + 1) & 1]);
    // B-fragment straight from LDS (row = output column, contiguous 32 K)
    v16bf bfr = *(const v16bf*)(&weff[c & 1][(wv * 16 + c16) * 32 + hh * 16]);
#pragma unroll
    for (int r = 0; r < 4; ++r) acc[r] = wmma_bf16(a[r], bfr, acc[r]);
    icur = icn; kcur = kn;
  }
  int col = o0 + wv * 16 + c16;
  float bv = bias_t[(size_t)n * outw + col];
  int rb = hh << 3;
  float* Ob = Out + (size_t)n * BBATCH * outw;
#pragma unroll
  for (int r = 0; r < 4; ++r)
#pragma unroll
    for (int jj = 0; jj < 8; ++jj)
      Ob[(size_t)(r*16 + rb + jj) * outw + col] = acc[r][jj] + bv;
}

// ---------------- elementwise / pack kernels ----------------
// Xg0[n,b,c] = XcT[b,c,n] = [ xt | (z?)*h | 0pad ]  (bf16, both layouts)
__global__ void k_pack(const float* __restrict__ X, int x_bs, int x_ns, int cx,
                       const float* __restrict__ Hst, const float* __restrict__ Zsc,
                       bf16_t* __restrict__ Xg0, bf16_t* __restrict__ XcT, int cin_p) {
  size_t idx = (size_t)blockIdx.x * blockDim.x + threadIdx.x;
  if (idx >= (size_t)NNODE * BBATCH * cin_p) return;
  int c = (int)(idx % cin_p); size_t nb = idx / cin_p;
  int b = (int)(nb % BBATCH); int n = (int)(nb / BBATCH);
  float v = 0.f;
  if (c < cx) {
    v = X[(size_t)b * x_bs + (size_t)n * x_ns + c];
  } else if (c < cx + HDIM) {
    int hh = c - cx;
    float hv = Hst[((size_t)n * BBATCH + b) * HDIM + hh];
    if (Zsc) hv *= Zsc[((size_t)n * BBATCH + b) * 256 + hh];
    v = hv;
  }
  bf16_t bv = (bf16_t)v;
  Xg0[idx] = bv;
  XcT[((size_t)b * cin_p + c) * NNODE + n] = bv;
}

__global__ void k_sigmoid(float* __restrict__ ZR, size_t ntot) {
  size_t idx = (size_t)blockIdx.x * blockDim.x + threadIdx.x;
  if (idx >= ntot) return;
  float x = ZR[idx];
  ZR[idx] = 1.f / (1.f + __expf(-x));
}

__global__ void k_update(const float* __restrict__ ZR, const float* __restrict__ HCr,
                         float* __restrict__ Hst, float* __restrict__ OutT, int t) {
  size_t idx = (size_t)blockIdx.x * blockDim.x + threadIdx.x;
  if (idx >= (size_t)NNODE * BBATCH * HDIM) return;
  int c = (int)(idx % HDIM); size_t nb = idx / HDIM;
  int b = (int)(nb % BBATCH); int n = (int)(nb / BBATCH);
  float r  = ZR[((size_t)n * BBATCH + b) * 256 + HDIM + c];
  float hc = tanhf(HCr[idx]);
  float h  = Hst[idx];
  float hn = r * h + (1.f - r) * hc;
  Hst[idx] = hn;
  OutT[(((size_t)b * TSTEP + t) * NNODE + n) * HDIM + c] = hn;
}

// ---------------- host driver ----------------
extern "C" void kernel_launch(void* const* d_in, const int* in_sizes, int n_in,
                              void* d_out, int out_size, void* d_ws, size_t ws_size,
                              hipStream_t stream) {
  (void)in_sizes; (void)n_in; (void)out_size; (void)ws_size;
  const float* src      = (const float*)d_in[0];
  const float* node_emb = (const float*)d_in[1];
  const float* time_emb = (const float*)d_in[2];
  const float* gW[2]   = {(const float*)d_in[3],  (const float*)d_in[11]};
  const float* gb[2]   = {(const float*)d_in[4],  (const float*)d_in[12]};
  const float* glng[2] = {(const float*)d_in[5],  (const float*)d_in[13]};
  const float* glnb[2] = {(const float*)d_in[6],  (const float*)d_in[14]};
  const float* uW[2]   = {(const float*)d_in[7],  (const float*)d_in[15]};
  const float* ub[2]   = {(const float*)d_in[8],  (const float*)d_in[16]};
  const float* ulng[2] = {(const float*)d_in[9],  (const float*)d_in[17]};
  const float* ulnb[2] = {(const float*)d_in[10], (const float*)d_in[18]};
  float* outp = (float*)d_out;

  const int cin_[2]  = {DIN_C + HDIM, 2 * HDIM};  // 130, 256
  const int cinp_[2] = {160, 256};                // zero-padded to 32-multiples
  const int cx_[2]   = {DIN_C, HDIM};

  char* ws = (char*)d_ws; size_t off = 0;
  auto wsa = [&](size_t bytes) -> void* {
    void* p = (void*)(ws + off);
    off += (bytes + 255) & ~(size_t)255;
    return p;
  };
  bf16_t* WtG[2]; bf16_t* WtU[2];
  WtG[0] = (bf16_t*)wsa((size_t)256 * 48 * 160 * 2);
  WtG[1] = (bf16_t*)wsa((size_t)256 * 48 * 256 * 2);
  WtU[0] = (bf16_t*)wsa((size_t)128 * 48 * 160 * 2);
  WtU[1] = (bf16_t*)wsa((size_t)128 * 48 * 256 * 2);
  float* neG   = (float*)wsa((size_t)TSTEP * NNODE * DEMB * 4);
  float* neU   = (float*)wsa((size_t)TSTEP * NNODE * DEMB * 4);
  float* biasG = (float*)wsa((size_t)TSTEP * NNODE * 256 * 4);
  float* biasU = (float*)wsa((size_t)TSTEP * NNODE * 128 * 4);
  bf16_t* Sb  = (bf16_t*)wsa((size_t)TSTEP * 2 * NNODE * NNODE * 2);
  bf16_t* Stb = (bf16_t*)wsa((size_t)TSTEP * 2 * NNODE * NNODE * 2);
  bf16_t* T2b = (bf16_t*)wsa((size_t)TSTEP * 2 * NNODE * NNODE * 2);
  bf16_t* Xg0 = (bf16_t*)wsa((size_t)NNODE * BBATCH * 256 * 2);
  bf16_t* XcT = (bf16_t*)wsa((size_t)BBATCH * 256 * NNODE * 2);
  bf16_t* Yb  = (bf16_t*)wsa((size_t)2 * NNODE * BBATCH * 256 * 2);
  float* ZR   = (float*)wsa((size_t)NNODE * BBATCH * 256 * 4);
  float* HC   = (float*)wsa((size_t)NNODE * BBATCH * 128 * 4);
  float* Hst  = (float*)wsa((size_t)NNODE * BBATCH * 128 * 4);

  // weight repack (per call; deterministic)
  for (int l = 0; l < 2; ++l) {
    size_t totg = (size_t)256 * 48 * cinp_[l];
    size_t totu = (size_t)128 * 48 * cinp_[l];
    k_wprep<<<dim3((unsigned)((totg + 255) / 256)), 256, 0, stream>>>(gW[l], WtG[l], cin_[l], cinp_[l], 256);
    k_wprep<<<dim3((unsigned)((totu + 255) / 256)), 256, 0, stream>>>(uW[l], WtU[l], cin_[l], cinp_[l], 128);
  }

  for (int l = 0; l < 2; ++l) {
    const int cp = cinp_[l];
    const int ncols = BBATCH * cp;            // 10240 (l0) / 16384 (l1)
    // per-layer, per-timestep embeddings, biases, graph supports
    k_neln<<<dim3((TSTEP * NNODE + 255) / 256), 256, 0, stream>>>(
        node_emb, time_emb, glng[l], glnb[l], ulng[l], ulnb[l], neG, neU);
    k_bias<<<dim3((unsigned)(((size_t)TSTEP * NNODE * 256 + 255) / 256)), 256, 0, stream>>>(neG, gb[l], biasG, 256);
    k_bias<<<dim3((unsigned)(((size_t)TSTEP * NNODE * 128 + 255) / 256)), 256, 0, stream>>>(neU, ub[l], biasU, 128);
    k_softmax<<<dim3(TSTEP * 2 * NNODE), 128, 0, stream>>>(neG, neU, Sb, Stb);
    k_t2<<<dim3(8, 2, TSTEP * 2), 256, 0, stream>>>(Sb, Stb, T2b);
    hipMemsetAsync(Hst, 0, (size_t)NNODE * BBATCH * HDIM * 4, stream);

    for (int t = 0; t < TSTEP; ++t) {
      const float* Xb; int xbs, xns;
      if (l == 0) { Xb = src  + (size_t)t * NNODE * DIN_C; xbs = TSTEP * NNODE * DIN_C; xns = DIN_C; }
      else        { Xb = outp + (size_t)t * NNODE * HDIM;  xbs = TSTEP * NNODE * HDIM;  xns = HDIM;  }
      size_t packtot = (size_t)NNODE * BBATCH * cp;
      unsigned packg = (unsigned)((packtot + 255) / 256);
      const bf16_t* Sg  = Sb  + (size_t)(t * 2 + 0) * NNODE * NNODE;
      const bf16_t* T2g = T2b + (size_t)(t * 2 + 0) * NNODE * NNODE;
      const bf16_t* Su  = Sb  + (size_t)(t * 2 + 1) * NNODE * NNODE;
      const bf16_t* T2u = T2b + (size_t)(t * 2 + 1) * NNODE * NNODE;

      // ---- gate GCN: zr = sigmoid(gcn([xt, h])) ----
      k_pack<<<dim3(packg), 256, 0, stream>>>(Xb, xbs, xns, cx_[l], Hst, (const float*)nullptr, Xg0, XcT, cp);
      k_gemm_ab<<<dim3(8, ncols / 256, 2), 256, 0, stream>>>(Sg, T2g, XcT, Yb, NNODE, ncols);
      k_gemm2<<<dim3(NNODE, 2), 256, 0, stream>>>(Xg0, Yb, WtG[l],
          neG + (size_t)t * NNODE * DEMB, biasG + (size_t)t * NNODE * 256, ZR, cp, 256);
      k_sigmoid<<<dim3((unsigned)(((size_t)NNODE * BBATCH * 256 + 255) / 256)), 256, 0, stream>>>(
          ZR, (size_t)NNODE * BBATCH * 256);

      // ---- update GCN: hc = tanh(gcn([xt, z*h])); h = r*h + (1-r)*hc ----
      k_pack<<<dim3(packg), 256, 0, stream>>>(Xb, xbs, xns, cx_[l], Hst, ZR, Xg0, XcT, cp);
      k_gemm_ab<<<dim3(8, ncols / 256, 2), 256, 0, stream>>>(Su, T2u, XcT, Yb, NNODE, ncols);
      k_gemm2<<<dim3(NNODE, 1), 256, 0, stream>>>(Xg0, Yb, WtU[l],
          neU + (size_t)t * NNODE * DEMB, biasU + (size_t)t * NNODE * 128, HC, cp, 128);
      k_update<<<dim3((unsigned)(((size_t)NNODE * BBATCH * HDIM + 255) / 256)), 256, 0, stream>>>(
          ZR, HC, Hst, outp, t);
    }
  }
}
